// SelfAttentiveLBLBiLM_45792941309970
// MI455X (gfx1250) — compile-verified, matrix operand order checked by reading, not run on hardware
//
#include <hip/hip_runtime.h>

typedef _Float16 half_t;
typedef __attribute__((ext_vector_type(16))) _Float16 v16h;
typedef __attribute__((ext_vector_type(8)))  _Float16 v8h;
typedef __attribute__((ext_vector_type(8)))  float    v8f;

namespace {
constexpr int   B_    = 8;
constexpr int   S_    = 1024;
constexpr int   D_    = 512;
constexpr int   H_    = 8;
constexpr int   DK_   = D_ / H_;          // 64
constexpr int   WIDTH_= 8;
constexpr int   NHW_  = 2;
constexpr int   S2_   = S_ + 2 * WIDTH_;  // 1040
constexpr long  NTOK_ = (long)B_ * S2_;   // 8320 tokens (padded)
constexpr long  NROW_ = (long)B_ * S_;    // 8192 tokens (unpadded)
constexpr int   WIN_  = WIDTH_ + 2;       // max 10 keys in band
}

// ---------------------------------------------------------------------------
// Weight convert + transpose: W f32 [K,N] (row = k)  ->  Wt f16 [N,K] (row = n)
// ---------------------------------------------------------------------------
__global__ void convert_wt_kernel(const float* __restrict__ W,
                                  half_t* __restrict__ Wt, int K, int N) {
    long idx = (long)blockIdx.x * blockDim.x + threadIdx.x;
    long total = (long)K * N;
    if (idx >= total) return;
    int  n = (int)(idx % N);
    long k = idx / N;
    Wt[(long)n * K + k] = (half_t)W[idx];
}

// ---------------------------------------------------------------------------
// Build new_in = [left_pad ; x ; right_pad], both f32 and f16 copies
// ---------------------------------------------------------------------------
__global__ void pad_concat_kernel(const float* __restrict__ x,
                                  const float* __restrict__ lp,
                                  const float* __restrict__ rp,
                                  float* __restrict__ nin,
                                  half_t* __restrict__ nin_h) {
    long idx = (long)blockIdx.x * blockDim.x + threadIdx.x;
    if (idx >= NTOK_ * D_) return;
    int  d  = (int)(idx % D_);
    long bs = idx / D_;
    int  s  = (int)(bs % S2_);
    int  b  = (int)(bs / S2_);
    float v;
    if (s < WIDTH_)            v = lp[(long)s * D_ + d];
    else if (s < WIDTH_ + S_)  v = x[((long)b * S_ + (s - WIDTH_)) * D_ + d];
    else                       v = rp[(long)(s - WIDTH_ - S_) * D_ + d];
    nin[idx]   = v;
    nin_h[idx] = (half_t)v;
}

// ---------------------------------------------------------------------------
// WMMA GEMM: C[M,N] = A[M,K] (f16, row-major) * Wt[N,K]^T (f16) + bias[N]
// BM=128, BN=64, BK=32; 256 threads = 8 wave32s, each wave owns 32x32 of C
// (4 accumulators, 4 WMMAs per K-step). Double-buffered LDS + register
// staging so global loads overlap the WMMAs; one barrier per K-step.
// M multiple of 128, N multiple of 64, K multiple of 32; 16B-aligned ptrs.
// ---------------------------------------------------------------------------
#define BM 128
#define BN 64
#define BK 32

__global__ __launch_bounds__(256) void gemm_wmma_kernel(
        const half_t* __restrict__ A, const half_t* __restrict__ Bt,
        const float* __restrict__ bias, float* __restrict__ C,
        int M, int N, int K) {
    __shared__ half_t As[2][BM][BK];   // 2 x 128 x 32 halves = 16 KB
    __shared__ half_t Bs[2][BN][BK];   // 2 x  64 x 32 halves =  8 KB

    const int tid  = threadIdx.x;
    const int lane = tid & 31;
    const int wave = tid >> 5;
    const int wr   = wave >> 1;     // 0..3 : 32-row block
    const int wc   = wave & 1;      // 0..1 : 32-col block

    const long rowBase = (long)blockIdx.x * BM;
    const long colBase = (long)blockIdx.y * BN;

    // A tile loader: 128 rows x 4 chunks(8 halves) = 512 chunks, 2 per thread
    const int ar0 = tid >> 2;               // chunk tid      -> row 0..63
    const int ar1 = (tid + 256) >> 2;       // chunk tid+256  -> row 64..127
    const int ac  = (tid & 3) * 8;          // 0,8,16,24
    // B tile loader: 64 rows x 4 chunks = 256 chunks, 1 per thread
    const int br  = tid >> 2;
    const int bc  = (tid & 3) * 8;

    // Fragment addressing per CDNA5 16-bit operand layout:
    // lane L: m = L%16, k-base = (L>>4)*8; elems [kb..kb+7] and [16+kb..16+kb+7]
    const int fm = lane & 15;
    const int kb = (lane >> 4) * 8;

    v8f acc00 = {}, acc01 = {}, acc10 = {}, acc11 = {};

    const half_t* pa0 = A  + (rowBase + ar0) * (long)K;
    const half_t* pa1 = A  + (rowBase + ar1) * (long)K;
    const half_t* pb  = Bt + (colBase + br ) * (long)K;

    // ---- prologue: stage tile 0 and commit to buffer 0 ----
    v8h sA0 = *(const v8h*)(pa0 + ac);
    v8h sA1 = *(const v8h*)(pa1 + ac);
    v8h sB  = *(const v8h*)(pb  + bc);
    *(v8h*)&As[0][ar0][ac] = sA0;
    *(v8h*)&As[0][ar1][ac] = sA1;
    *(v8h*)&Bs[0][br ][bc] = sB;
    __syncthreads();

    const int nIter = K / BK;
    int buf = 0;
    for (int it = 0; it < nIter; ++it) {
        const bool more = (it + 1) < nIter;
        if (more) {                      // issue next tile's loads early
            const int ko = (it + 1) * BK;
            sA0 = *(const v8h*)(pa0 + ko + ac);
            sA1 = *(const v8h*)(pa1 + ko + ac);
            sB  = *(const v8h*)(pb  + ko + bc);
        }

        v16h a0, a1, b0, b1;
        *((v8h*)&a0)     = *(const v8h*)&As[buf][wr * 32 + fm][kb];
        *((v8h*)&a0 + 1) = *(const v8h*)&As[buf][wr * 32 + fm][16 + kb];
        *((v8h*)&a1)     = *(const v8h*)&As[buf][wr * 32 + 16 + fm][kb];
        *((v8h*)&a1 + 1) = *(const v8h*)&As[buf][wr * 32 + 16 + fm][16 + kb];
        *((v8h*)&b0)     = *(const v8h*)&Bs[buf][wc * 32 + fm][kb];
        *((v8h*)&b0 + 1) = *(const v8h*)&Bs[buf][wc * 32 + fm][16 + kb];
        *((v8h*)&b1)     = *(const v8h*)&Bs[buf][wc * 32 + 16 + fm][kb];
        *((v8h*)&b1 + 1) = *(const v8h*)&Bs[buf][wc * 32 + 16 + fm][16 + kb];

        acc00 = __builtin_amdgcn_wmma_f32_16x16x32_f16(false, a0, false, b0,
                                                       (short)0, acc00, false, false);
        acc01 = __builtin_amdgcn_wmma_f32_16x16x32_f16(false, a0, false, b1,
                                                       (short)0, acc01, false, false);
        acc10 = __builtin_amdgcn_wmma_f32_16x16x32_f16(false, a1, false, b0,
                                                       (short)0, acc10, false, false);
        acc11 = __builtin_amdgcn_wmma_f32_16x16x32_f16(false, a1, false, b1,
                                                       (short)0, acc11, false, false);

        if (more) {                      // commit staged tile to other buffer
            *(v8h*)&As[buf ^ 1][ar0][ac] = sA0;
            *(v8h*)&As[buf ^ 1][ar1][ac] = sA1;
            *(v8h*)&Bs[buf ^ 1][br ][bc] = sB;
        }
        __syncthreads();                 // one barrier per K-step
        buf ^= 1;
    }

    // C layout: VGPR r -> row = r + 8*(lane>>4), col = lane&15 (per 16-strip)
    const int  n0 = lane & 15;
    const int  mo = (lane >> 4) * 8;
    const long rA = rowBase + wr * 32 + mo;        // rows of acc0*
    const long rB = rA + 16;                       // rows of acc1*
    const long c0 = colBase + wc * 32 + n0;
    const float bv0 = bias[c0];
    const float bv1 = bias[c0 + 16];
#pragma unroll
    for (int r = 0; r < 8; ++r) {
        C[(rA + r) * N + c0]      = acc00[r] + bv0;
        C[(rA + r) * N + c0 + 16] = acc01[r] + bv1;
        C[(rB + r) * N + c0]      = acc10[r] + bv0;
        C[(rB + r) * N + c0 + 16] = acc11[r] + bv1;
    }
}

// ---------------------------------------------------------------------------
// Banded attention: one thread per (side, b, h, s). <=10 keys per query.
// qkv = 6 consecutive [NTOK_*D_] f32 blocks: (side*3 + {q,k,v}).
// ctx = 2 consecutive [NROW_*D_] f16 blocks (A operand of O-projection).
// ---------------------------------------------------------------------------
__global__ __launch_bounds__(256) void win_attn_kernel(
        const float* __restrict__ qkv, half_t* __restrict__ ctx) {
    long t = (long)blockIdx.x * blockDim.x + threadIdx.x;
    const long NQ = (long)B_ * H_ * S_;
    if (t >= 2 * NQ) return;
    const int side = (int)(t / NQ);
    long r = t % NQ;
    const int s = (int)(r % S_); r /= S_;
    const int h = (int)(r % H_);
    const int b = (int)(r / H_);

    const float* q = qkv + ((long)side * 3 + 0) * NTOK_ * D_;
    const float* k = qkv + ((long)side * 3 + 1) * NTOK_ * D_;
    const float* v = qkv + ((long)side * 3 + 2) * NTOK_ * D_;

    const int i = WIDTH_ + s;
    int jlo, jhi;
    if (side == 0) { jlo = i - (WIDTH_ + 1); if (jlo < 0) jlo = 0; jhi = i; }
    else           { jlo = i; jhi = i + (WIDTH_ + 1); if (jhi > S2_ - 1) jhi = S2_ - 1; }
    const int cnt = jhi - jlo + 1;

    float qr[DK_];
    const float* qp = q + ((long)b * S2_ + i) * D_ + (long)h * DK_;
#pragma unroll
    for (int d = 0; d < DK_; ++d) qr[d] = qp[d];

    float sc[WIN_];
    float mx = -1e30f;
    for (int j = 0; j < cnt; ++j) {
        const float* kp = k + ((long)b * S2_ + jlo + j) * D_ + (long)h * DK_;
        float a = 0.f;
#pragma unroll
        for (int d = 0; d < DK_; ++d) a += qr[d] * kp[d];
        a *= 0.125f;                       // 1/sqrt(64)
        sc[j] = a;
        if (a > mx) mx = a;
    }
    float den = 0.f;
    for (int j = 0; j < cnt; ++j) { sc[j] = __expf(sc[j] - mx); den += sc[j]; }
    const float inv = 1.f / den;

    half_t* op = ctx + ((long)side * NROW_ + (long)b * S_ + s) * D_ + (long)h * DK_;
    const float* vb = v + ((long)b * S2_ + jlo) * D_ + (long)h * DK_;
#pragma unroll 4
    for (int d = 0; d < DK_; ++d) {
        float o = 0.f;
        for (int j = 0; j < cnt; ++j) o += sc[j] * vb[(long)j * D_ + d];
        op[d] = (half_t)(o * inv);
    }
}

// ---------------------------------------------------------------------------
// x = oproj_out + sum_j wrel[j] * new_in[b, baseOff + j + s, :]; write f32+f16
// ---------------------------------------------------------------------------
__global__ void add_rel_kernel(const float* __restrict__ oproj,
                               const float* __restrict__ nin,
                               const float* __restrict__ wrel, int baseOff,
                               float* __restrict__ x, half_t* __restrict__ xh) {
    long idx = (long)blockIdx.x * blockDim.x + threadIdx.x;
    if (idx >= NROW_ * D_) return;
    const int  d  = (int)(idx % D_);
    long bs = idx / D_;
    const int  s  = (int)(bs % S_);
    const int  b  = (int)(bs / S_);
    float acc = oproj[idx];
#pragma unroll
    for (int j = 0; j <= WIDTH_; ++j)
        acc += wrel[j] * nin[((long)b * S2_ + baseOff + j + s) * D_ + d];
    x[idx]  = acc;
    xh[idx] = (half_t)acc;
}

// ---------------------------------------------------------------------------
// Highway gating: proj[B*S,2D] (bias added) -> x = g*x + (1-g)*relu(nl)
// On last layer also writes the output slice (stride 2D, offset outOff).
// ---------------------------------------------------------------------------
__global__ void highway_ew_kernel(const float* __restrict__ proj,
                                  float* __restrict__ x, half_t* __restrict__ xh,
                                  float* __restrict__ out_final, int outOff) {
    long idx = (long)blockIdx.x * blockDim.x + threadIdx.x;
    if (idx >= NROW_ * D_) return;
    const long row = idx / D_;
    const int  d   = (int)(idx % D_);
    float nl = proj[row * (2 * D_) + d];
    nl = nl > 0.f ? nl : 0.f;
    const float gv = proj[row * (2 * D_) + D_ + d];
    const float g  = 1.f / (1.f + __expf(-gv));
    const float xn = g * x[idx] + (1.f - g) * nl;
    x[idx]  = xn;
    xh[idx] = (half_t)xn;
    if (out_final) out_final[row * (2 * D_) + outOff + d] = xn;
}

// ---------------------------------------------------------------------------
extern "C" void kernel_launch(void* const* d_in, const int* in_sizes, int n_in,
                              void* d_out, int out_size, void* d_ws, size_t ws_size,
                              hipStream_t stream) {
    (void)in_sizes; (void)n_in; (void)out_size; (void)ws_size;

    const float* x_in     = (const float*)d_in[0];
    const float* l_attn_W = (const float*)d_in[1];
    const float* l_attn_b = (const float*)d_in[2];
    const float* r_attn_W = (const float*)d_in[3];
    const float* r_attn_b = (const float*)d_in[4];
    const float* left_pad = (const float*)d_in[5];
    const float* right_pad= (const float*)d_in[6];
    const float* left_w   = (const float*)d_in[7];
    const float* right_w  = (const float*)d_in[8];
    const float* l_hw_W   = (const float*)d_in[9];
    const float* l_hw_b   = (const float*)d_in[10];
    const float* r_hw_W   = (const float*)d_in[11];
    const float* r_hw_b   = (const float*)d_in[12];
    float* out = (float*)d_out;

    // ---- workspace layout (256B aligned carve) ----
    char* ws = (char*)d_ws;
    size_t off = 0;
    auto carve = [&](size_t bytes) -> char* {
        char* p = ws + off;
        off += (bytes + 255) & ~(size_t)255;
        return p;
    };
    float*  nin    = (float*) carve(NTOK_ * D_ * sizeof(float));
    half_t* nin_h  = (half_t*)carve(NTOK_ * D_ * sizeof(half_t));
    // transposed f16 weights: 8 attn [D,D] then 2*NHW highway [2D,D]
    const long ATT_W  = (long)D_ * D_;         // halves per attn matrix
    const long HW_W   = (long)2 * D_ * D_;     // halves per highway matrix
    half_t* wbuf   = (half_t*)carve((8 * ATT_W + 2 * NHW_ * HW_W) * sizeof(half_t));
    float*  qkv    = (float*) carve(6 * NTOK_ * D_ * sizeof(float));
    half_t* ctx    = (half_t*)carve(2 * NROW_ * D_ * sizeof(half_t));
    float*  tmp    = (float*) carve(NROW_ * D_ * sizeof(float));
    float*  xbuf   = (float*) carve(NROW_ * D_ * sizeof(float));
    half_t* xh     = (half_t*)carve(NROW_ * D_ * sizeof(half_t));
    float*  proj   = (float*) carve(NROW_ * 2 * D_ * sizeof(float));

    const int TPB = 256;

    // ---- 1) convert + transpose weights to f16 [N][K] ----
    {
        dim3 g((unsigned)(((long)D_ * D_ + TPB - 1) / TPB));
        for (int i = 0; i < 4; ++i) {
            convert_wt_kernel<<<g, TPB, 0, stream>>>(l_attn_W + (long)i * D_ * D_,
                                                     wbuf + i * ATT_W, D_, D_);
            convert_wt_kernel<<<g, TPB, 0, stream>>>(r_attn_W + (long)i * D_ * D_,
                                                     wbuf + (4 + i) * ATT_W, D_, D_);
        }
        dim3 gh((unsigned)(((long)D_ * 2 * D_ + TPB - 1) / TPB));
        for (int i = 0; i < NHW_; ++i) {
            convert_wt_kernel<<<gh, TPB, 0, stream>>>(l_hw_W + (long)i * D_ * 2 * D_,
                    wbuf + 8 * ATT_W + i * HW_W, D_, 2 * D_);
            convert_wt_kernel<<<gh, TPB, 0, stream>>>(r_hw_W + (long)i * D_ * 2 * D_,
                    wbuf + 8 * ATT_W + (NHW_ + i) * HW_W, D_, 2 * D_);
        }
    }

    // ---- 2) pad/concat ----
    {
        dim3 g((unsigned)((NTOK_ * D_ + TPB - 1) / TPB));
        pad_concat_kernel<<<g, TPB, 0, stream>>>(x_in, left_pad, right_pad, nin, nin_h);
    }

    // ---- 3) QKV projections (6 WMMA GEMMs, M=8320, N=K=512) ----
    {
        dim3 g((unsigned)(NTOK_ / BM), (unsigned)(D_ / BN));
        for (int side = 0; side < 2; ++side) {
            const float* ab = side ? r_attn_b : l_attn_b;
            for (int j = 0; j < 3; ++j) {
                gemm_wmma_kernel<<<g, TPB, 0, stream>>>(
                    nin_h, wbuf + ((long)side * 4 + j) * ATT_W, ab + (long)j * D_,
                    qkv + ((long)side * 3 + j) * NTOK_ * D_,
                    (int)NTOK_, D_, D_);
            }
        }
    }

    // ---- 4) banded attention (both sides) ----
    {
        dim3 g((unsigned)((2L * B_ * H_ * S_ + TPB - 1) / TPB));
        win_attn_kernel<<<g, TPB, 0, stream>>>(qkv, ctx);
    }

    // ---- 5) per side: O-proj GEMM, +rel, highway x2 ----
    for (int side = 0; side < 2; ++side) {
        const float* ab = side ? r_attn_b : l_attn_b;
        const float* hb = side ? r_hw_b : l_hw_b;
        const float* wr = side ? right_w : left_w;
        const int baseOff = side ? WIDTH_ : 0;

        {   // O-projection: [8192,512] x [512,512]
            dim3 g((unsigned)(NROW_ / BM), (unsigned)(D_ / BN));
            gemm_wmma_kernel<<<g, TPB, 0, stream>>>(
                ctx + (long)side * NROW_ * D_,
                wbuf + ((long)side * 4 + 3) * ATT_W, ab + 3L * D_,
                tmp, (int)NROW_, D_, D_);
        }
        {   // + relative-position mix -> x (f32 + f16)
            dim3 g((unsigned)((NROW_ * D_ + TPB - 1) / TPB));
            add_rel_kernel<<<g, TPB, 0, stream>>>(tmp, nin, wr, baseOff, xbuf, xh);
        }
        for (int l = 0; l < NHW_; ++l) {
            {   // highway proj: [8192,512] x [512,1024]
                dim3 g((unsigned)(NROW_ / BM), (unsigned)(2 * D_ / BN));
                gemm_wmma_kernel<<<g, TPB, 0, stream>>>(
                    xh, wbuf + 8 * ATT_W + ((long)side * NHW_ + l) * HW_W,
                    hb + (long)l * 2 * D_, proj, (int)NROW_, 2 * D_, D_);
            }
            {   // gating; last layer streams into d_out slice
                dim3 g((unsigned)((NROW_ * D_ + TPB - 1) / TPB));
                highway_ew_kernel<<<g, TPB, 0, stream>>>(
                    proj, xbuf, xh,
                    (l == NHW_ - 1) ? out : (float*)nullptr, side * D_);
            }
        }
    }
}
